// MolPCBANet_10007273799857
// MI455X (gfx1250) — matrix operand order repaired and checked by reading, not compiled, not run on hardware
//
#include <hip/hip_runtime.h>
#include <hip/hip_bf16.h>

// ---------------------------------------------------------------------------
// MolPCBA GIN-VirtualNode forward on gfx1250 (MI455X).
//
// GEMMs: V_WMMA_F32_16X16X32_F16, f32 accumulate. Weights are pre-swizzled
// (one-time device kernels) into the exact per-lane WMMA B-fragment layout
// as f16, so the hot loop is pure b128 loads + wmma with zero guards.
// Activation buffers use zero-padded leading dims (300->320, 600->608) so
// the K loop needs no bounds checks either. Scatter/gather phases
// (edge aggregation, segment sums, mean pool) ride on global f32 atomics.
// ---------------------------------------------------------------------------

typedef __attribute__((ext_vector_type(16))) _Float16 v16h;
typedef __attribute__((ext_vector_type(8)))  _Float16 v8h;
typedef __attribute__((ext_vector_type(8)))  float    v8f;

#define NN 100000
#define EE 250000
#define GG 4096
#define DD 300
#define LL 3
#define TT 128
#define D2 (2*DD)

// padded (multiples of 32) leading dims for activations / K dims
#define DP  320
#define D2P 608

// 1/sqrt(1 + 1e-5) -- eval-mode BN with running stats (0,1)
#define BN_INV 0.9999950000374997f

// ---------------- one-time weight swizzle into WMMA B-fragment order -------
// dst layout: [Kpad/32][Npad/16][lane 0..31][elem 0..15] f16, so a GEMM wave
// reads its whole 32x16 B fragment as 32 contiguous bytes per lane.
// Fragment semantics: half=lane>>4; elem i<8 -> K=c*32+half*8+i, i>=8 ->
// K=c*32+16+half*8+(i-8);  n = tile*16 + (lane&15).
__global__ void swizzle_w_kernel(const float* __restrict__ W, int K, int Nc,
                                 int Kpad, int Npad, _Float16* __restrict__ dst) {
  long idx = (long)blockIdx.x * blockDim.x + threadIdx.x;
  long total = (long)Kpad * Npad;
  if (idx >= total) return;
  int i    = (int)(idx & 15);
  int lane = (int)((idx >> 4) & 31);
  long rest = idx >> 9;
  int tilesN = Npad >> 4;
  int t = (int)(rest % tilesN);
  int c = (int)(rest / tilesN);
  int k = c * 32 + ((lane >> 4) << 3) + ((i >> 3) << 4) + (i & 7);
  int n = (t << 4) + (lane & 15);
  float v = (k < K && n < Nc) ? W[(long)k * Nc + n] : 0.0f;
  dst[idx] = (_Float16)v;
}

// ---------------- elementwise / gather-scatter kernels ---------------------

__global__ void atom_encode_kernel(const int* __restrict__ x,
                                   const float* __restrict__ atom_emb,
                                   float* __restrict__ h) {
  long idx = (long)blockIdx.x * blockDim.x + threadIdx.x;
  if (idx >= (long)NN * DD) return;
  int n = (int)(idx / DD), d = (int)(idx % DD);
  float s = 0.0f;
#pragma unroll
  for (int f = 0; f < 9; ++f) {
    int cat = x[n * 9 + f];
    s += atom_emb[((long)f * 64 + cat) * DD + d];
  }
  h[(long)n * DP + d] = s;
}

__global__ void vn_init_kernel(const float* __restrict__ vn_emb,
                               float* __restrict__ vn) {
  long idx = (long)blockIdx.x * blockDim.x + threadIdx.x;
  if (idx >= (long)GG * DP) return;
  int d = (int)(idx % DP);
  vn[idx] = (d < DD) ? vn_emb[d] : 0.0f;   // keep pad columns zero
}

__global__ void add_vn_kernel(const float* __restrict__ h,
                              const float* __restrict__ vn,
                              const int* __restrict__ batch,
                              float* __restrict__ h_in) {
  long idx = (long)blockIdx.x * blockDim.x + threadIdx.x;
  if (idx >= (long)NN * DD) return;
  int n = (int)(idx / DD), d = (int)(idx % DD);
  h_in[(long)n * DP + d] = h[(long)n * DP + d] + vn[(long)batch[n] * DP + d];
}

// msg = relu(h_in[src] + bond_enc); aggr[dst] += msg   (coalesced over d)
__global__ void edge_msg_kernel(const int* __restrict__ edge_index,
                                const int* __restrict__ edge_attr,
                                const float* __restrict__ bond_emb_l,
                                const float* __restrict__ h_in,
                                float* __restrict__ aggr) {
  long idx = (long)blockIdx.x * blockDim.x + threadIdx.x;
  if (idx >= (long)EE * DD) return;
  int e = (int)(idx / DD), d = (int)(idx % DD);
  int src = edge_index[e];         // row 0
  int dst = edge_index[EE + e];    // row 1
  float bv = 0.0f;
#pragma unroll
  for (int f = 0; f < 3; ++f) {
    int cat = edge_attr[e * 3 + f];
    bv += bond_emb_l[((long)f * 8 + cat) * DD + d];
  }
  float msg = fmaxf(h_in[(long)src * DP + d] + bv, 0.0f);
  atomicAdd(&aggr[(long)dst * DP + d], msg);
}

// z = (1+eps[l])*h_in + aggr, written as f16 (GEMM1 input). Real cols only;
// pad columns of z16 were zeroed once and are never touched again.
__global__ void make_z_kernel(const float* __restrict__ h_in,
                              const float* __restrict__ aggr,
                              const float* __restrict__ eps, int l,
                              _Float16* __restrict__ z16) {
  long idx = (long)blockIdx.x * blockDim.x + threadIdx.x;
  if (idx >= (long)NN * DD) return;
  int n = (int)(idx / DD), d = (int)(idx % DD);
  long o = (long)n * DP + d;
  z16[o] = (_Float16)fmaf(1.0f + eps[l], h_in[o], aggr[o]);
}

// dst_g[batch[n]] += src[n]  over real columns
__global__ void seg_add_kernel(const float* __restrict__ src,
                               const int* __restrict__ batch,
                               float* __restrict__ dst_g) {
  long idx = (long)blockIdx.x * blockDim.x + threadIdx.x;
  if (idx >= (long)NN * DD) return;
  int n = (int)(idx / DD), d = (int)(idx % DD);
  atomicAdd(&dst_g[(long)batch[n] * DP + d], src[(long)n * DP + d]);
}

__global__ void count_kernel(const int* __restrict__ batch,
                             float* __restrict__ cnt) {
  int n = blockIdx.x * blockDim.x + threadIdx.x;
  if (n >= NN) return;
  atomicAdd(&cnt[batch[n]], 1.0f);
}

__global__ void div_kernel(float* __restrict__ pool,
                           const float* __restrict__ cnt) {
  long idx = (long)blockIdx.x * blockDim.x + threadIdx.x;
  if (idx >= (long)GG * DP) return;
  pool[idx] /= fmaxf(cnt[idx / DP], 1.0f);   // 0/x = 0 keeps pads zero
}

// ---------------- WMMA GEMM ------------------------------------------------
// A-fragment loaders (per documented 16-bit A 16x32 layout: lane halves carry
// K-halves; elem i<8 -> K = base+i, i>=8 -> K = base+16+(i-8), base = half*8).

__device__ inline v16h load_a_frag(const float* __restrict__ Ab, int k0) {
  const float4* p = (const float4*)(Ab + k0);        // K base .. base+7
  const float4* q = (const float4*)(Ab + k0 + 16);   // K base+16 .. base+23
  float4 a0 = p[0], a1 = p[1], a2 = q[0], a3 = q[1];
  v16h a;
  a[0] = (_Float16)a0.x; a[1] = (_Float16)a0.y; a[2] = (_Float16)a0.z; a[3] = (_Float16)a0.w;
  a[4] = (_Float16)a1.x; a[5] = (_Float16)a1.y; a[6] = (_Float16)a1.z; a[7] = (_Float16)a1.w;
  a[8] = (_Float16)a2.x; a[9] = (_Float16)a2.y; a[10] = (_Float16)a2.z; a[11] = (_Float16)a2.w;
  a[12] = (_Float16)a3.x; a[13] = (_Float16)a3.y; a[14] = (_Float16)a3.z; a[15] = (_Float16)a3.w;
  return a;
}

__device__ inline v16h load_a_frag(const _Float16* __restrict__ Ab, int k0) {
  v8h lo = *(const v8h*)(Ab + k0);        // K base .. base+7
  v8h hi = *(const v8h*)(Ab + k0 + 16);   // K base+16 .. base+23
  v16h a;
#pragma unroll
  for (int i = 0; i < 8; ++i) { a[i] = lo[i]; a[i + 8] = hi[i]; }
  return a;
}

// C[M,Npad] = act( bn( A[M,Kpad] @ Bswz + bias ) ).  One wave computes a
// 16x32 C strip (2 N-tiles, A fragment reused). No guards in the K loop:
// A pad columns are zero, B pads were zeroed by the swizzle. Epilogue writes
// 0 to pad columns (n >= NcReal) so padded buffers stay clean.
template <typename AT, typename CT>
__global__ __launch_bounds__(256)
void gemm_wmma_kernel(const AT* __restrict__ A, int lda,
                      const _Float16* __restrict__ Bs,
                      CT* __restrict__ C, int ldc,
                      int Kpad, int Npad, int NcReal,
                      const float* __restrict__ bias,   // per-col or null
                      const float* __restrict__ bn_g,   // per-col or null
                      const float* __restrict__ bn_b,   // per-col or null
                      int do_relu) {
  const int lane   = threadIdx.x & 31;
  const int wave   = threadIdx.x >> 5;
  const int tileM  = blockIdx.x;
  const int tilesN = Npad >> 4;
  const int tileN0 = (blockIdx.y * 8 + wave) << 1;
  if (tileN0 >= tilesN) return;            // whole-wave exit: EXEC stays full

  const int half = lane >> 4;
  const int m    = (tileM << 4) + (lane & 15);
  const int nloc = lane & 15;

  const AT* Abase = A + (size_t)m * lda + (half << 3);
  const _Float16* B0 = Bs + (((size_t)tileN0 * 32 + lane) << 4);
  const _Float16* B1 = B0 + (32 << 4);
  const size_t bstride = (size_t)tilesN * 512;   // f16 elems per K-chunk

  v8f acc0 = {}, acc1 = {};
  for (int k0 = 0; k0 < Kpad; k0 += 32) {
    v16h a  = load_a_frag(Abase, k0);
    v16h b0 = *(const v16h*)B0;
    v16h b1 = *(const v16h*)B1;
    __builtin_prefetch((const void*)(B0 + bstride), 0, 0);
    B0 += bstride; B1 += bstride;
    acc0 = __builtin_amdgcn_wmma_f32_16x16x32_f16(false, a, false, b0,
                                                  (short)0, acc0, false, false);
    acc1 = __builtin_amdgcn_wmma_f32_16x16x32_f16(false, a, false, b1,
                                                  (short)0, acc1, false, false);
  }

#pragma unroll
  for (int j = 0; j < 2; ++j) {
    const v8f& acc = j ? acc1 : acc0;
    int n = ((tileN0 + j) << 4) + nloc;
    bool live = n < NcReal;
    float bb = (live && bias) ? bias[n] : 0.0f;
    float s = 1.0f, t = 0.0f;
    if (live && bn_g) { s = bn_g[n] * BN_INV; t = bn_b[n]; }
    // C/D layout: VGPR r -> row r (lanes 0-15) / row r+8 (lanes 16-31)
#pragma unroll
    for (int r = 0; r < 8; ++r) {
      int row = (tileM << 4) + r + (half << 3);
      float v = (acc[r] + bb) * s + t;
      if (do_relu) v = fmaxf(v, 0.0f);
      if (!live) v = 0.0f;                  // keep pad columns zero
      C[(size_t)row * ldc + n] = (CT)v;
    }
  }
}

// ---------------------------------------------------------------------------

static inline dim3 gemm_grid(int M, int Npad) {
  int pairs = (Npad >> 4) >> 1;
  return dim3((unsigned)(M / 16), (unsigned)((pairs + 7) / 8));
}

extern "C" void kernel_launch(void* const* d_in, const int* in_sizes, int n_in,
                              void* d_out, int out_size, void* d_ws, size_t ws_size,
                              hipStream_t stream) {
  const int*   x          = (const int*)  d_in[0];
  const int*   edge_index = (const int*)  d_in[1];
  const int*   edge_attr  = (const int*)  d_in[2];
  const int*   batch      = (const int*)  d_in[3];
  const float* atom_emb   = (const float*)d_in[4];
  const float* bond_emb   = (const float*)d_in[5];
  const float* eps        = (const float*)d_in[6];
  const float* mlp_W1     = (const float*)d_in[7];
  const float* mlp_b1     = (const float*)d_in[8];
  const float* mlp_bn1_g  = (const float*)d_in[9];
  const float* mlp_bn1_b  = (const float*)d_in[10];
  const float* mlp_W2     = (const float*)d_in[11];
  const float* mlp_b2     = (const float*)d_in[12];
  const float* bn_g       = (const float*)d_in[13];
  const float* bn_b       = (const float*)d_in[14];
  const float* vn_W1      = (const float*)d_in[15];
  const float* vn_b1      = (const float*)d_in[16];
  const float* vn_bn1_g   = (const float*)d_in[17];
  const float* vn_bn1_b   = (const float*)d_in[18];
  const float* vn_W2      = (const float*)d_in[19];
  const float* vn_b2      = (const float*)d_in[20];
  const float* vn_bn2_g   = (const float*)d_in[21];
  const float* vn_bn2_b   = (const float*)d_in[22];
  const float* vn_emb     = (const float*)d_in[23];
  const float* causal_W   = (const float*)d_in[24];
  const float* causal_b   = (const float*)d_in[25];
  float* out = (float*)d_out;
  (void)in_sizes; (void)n_in; (void)out_size; (void)ws_size;

  // ---- workspace carve-out ----
  char* wsp = (char*)d_ws;
  auto carveB = [&](size_t bytes) -> void* {
    void* p = (void*)wsp;
    wsp += (bytes + 255) & ~(size_t)255;
    return p;
  };
  float*     H      = (float*)    carveB((size_t)NN * DP  * 4);  // node features
  float*     HIN    = (float*)    carveB((size_t)NN * DP  * 4);  // h + vn[batch]
  float*     AGGR   = (float*)    carveB((size_t)NN * DP  * 4);  // edge aggregation (f32, atomics)
  _Float16*  Z16    = (_Float16*) carveB((size_t)NN * DP  * 2);  // GEMM1 input (f16)
  _Float16*  T1     = (_Float16*) carveB((size_t)NN * D2P * 2);  // node-MLP hidden (f16)
  float*     VN     = (float*)    carveB((size_t)GG * DP  * 4);  // virtual node state
  float*     VT     = (float*)    carveB((size_t)GG * DP  * 4);  // vn segment-sum input
  _Float16*  VT1    = (_Float16*) carveB((size_t)GG * D2P * 2);  // vn-MLP hidden (f16)
  float*     POOL   = (float*)    carveB((size_t)GG * DP  * 4);  // mean-pool accumulator
  float*     CNT    = (float*)    carveB((size_t)GG * 4);
  _Float16*  W1S    = (_Float16*) carveB((size_t)LL * DP * D2P * 2);       // swizzled weights
  _Float16*  W2S    = (_Float16*) carveB((size_t)LL * D2P * DP * 2);
  _Float16*  V1S    = (_Float16*) carveB((size_t)(LL-1) * DP * D2P * 2);
  _Float16*  V2S    = (_Float16*) carveB((size_t)(LL-1) * D2P * DP * 2);
  _Float16*  CWS    = (_Float16*) carveB((size_t)DP * TT * 2);

  const long nd = (long)NN * DD;
  const long ed = (long)EE * DD;
  const int  B = 256;
  auto blocks = [](long n) { return (unsigned)((n + 255) / 256); };

  // ---- one-time weight swizzles (f32 -> f16 WMMA B-fragment order) ----
  for (int l = 0; l < LL; ++l) {
    swizzle_w_kernel<<<blocks((long)DP * D2P), B, 0, stream>>>(
        mlp_W1 + (size_t)l * DD * D2, DD, D2, DP, D2P, W1S + (size_t)l * DP * D2P);
    swizzle_w_kernel<<<blocks((long)D2P * DP), B, 0, stream>>>(
        mlp_W2 + (size_t)l * D2 * DD, D2, DD, D2P, DP, W2S + (size_t)l * D2P * DP);
  }
  for (int l = 0; l < LL - 1; ++l) {
    swizzle_w_kernel<<<blocks((long)DP * D2P), B, 0, stream>>>(
        vn_W1 + (size_t)l * DD * D2, DD, D2, DP, D2P, V1S + (size_t)l * DP * D2P);
    swizzle_w_kernel<<<blocks((long)D2P * DP), B, 0, stream>>>(
        vn_W2 + (size_t)l * D2 * DD, D2, DD, D2P, DP, V2S + (size_t)l * D2P * DP);
  }
  swizzle_w_kernel<<<blocks((long)DP * TT), B, 0, stream>>>(
      causal_W, DD, TT, DP, TT, CWS);

  // ---- zero pad columns of Z16 once (make_z only writes real columns) ----
  hipMemsetAsync(Z16, 0, (size_t)NN * DP * 2, stream);

  atom_encode_kernel<<<blocks(nd), B, 0, stream>>>(x, atom_emb, H);
  vn_init_kernel<<<blocks((long)GG * DP), B, 0, stream>>>(vn_emb, VN);

  for (int l = 0; l < LL; ++l) {
    add_vn_kernel<<<blocks(nd), B, 0, stream>>>(H, VN, batch, HIN);

    hipMemsetAsync(AGGR, 0, (size_t)NN * DP * 4, stream);
    edge_msg_kernel<<<blocks(ed), B, 0, stream>>>(
        edge_index, edge_attr, bond_emb + (size_t)l * 3 * 8 * DD, HIN, AGGR);

    make_z_kernel<<<blocks(nd), B, 0, stream>>>(HIN, AGGR, eps, l, Z16);

    // GEMM1: T1 = relu(bn1(z @ W1 + b1))   [100000x300]x[300x600] (f16 in/out)
    gemm_wmma_kernel<_Float16, _Float16><<<gemm_grid(NN, D2P), 256, 0, stream>>>(
        Z16, DP, W1S + (size_t)l * DP * D2P, T1, D2P,
        DP, D2P, D2,
        mlp_b1 + (size_t)l * D2, mlp_bn1_g + (size_t)l * D2, mlp_bn1_b + (size_t)l * D2, 1);

    // GEMM2: H = bn(T1 @ W2 + b2) (+relu except last layer)
    gemm_wmma_kernel<_Float16, float><<<gemm_grid(NN, DP), 256, 0, stream>>>(
        T1, D2P, W2S + (size_t)l * D2P * DP, H, DP,
        D2P, DP, DD,
        mlp_b2 + (size_t)l * DD, bn_g + (size_t)l * DD, bn_b + (size_t)l * DD,
        (l < LL - 1) ? 1 : 0);

    if (l < LL - 1) {
      // vn_tmp = segment_sum(h_in, batch) + vn
      hipMemcpyAsync(VT, VN, (size_t)GG * DP * 4, hipMemcpyDeviceToDevice, stream);
      seg_add_kernel<<<blocks(nd), B, 0, stream>>>(HIN, batch, VT);

      gemm_wmma_kernel<float, _Float16><<<gemm_grid(GG, D2P), 256, 0, stream>>>(
          VT, DP, V1S + (size_t)l * DP * D2P, VT1, D2P,
          DP, D2P, D2,
          vn_b1 + (size_t)l * D2, vn_bn1_g + (size_t)l * D2, vn_bn1_b + (size_t)l * D2, 1);
      gemm_wmma_kernel<_Float16, float><<<gemm_grid(GG, DP), 256, 0, stream>>>(
          VT1, D2P, V2S + (size_t)l * D2P * DP, VN, DP,
          D2P, DP, DD,
          vn_b2 + (size_t)l * DD, vn_bn2_g + (size_t)l * DD, vn_bn2_b + (size_t)l * DD, 1);
    }
  }

  // global mean pool
  hipMemsetAsync(POOL, 0, (size_t)GG * DP * 4, stream);
  hipMemsetAsync(CNT, 0, (size_t)GG * 4, stream);
  count_kernel<<<(NN + B - 1) / B, B, 0, stream>>>(batch, CNT);
  seg_add_kernel<<<blocks(nd), B, 0, stream>>>(H, batch, POOL);
  div_kernel<<<blocks((long)GG * DP), B, 0, stream>>>(POOL, CNT);

  // classifier: out = POOL @ causal_W + causal_b   [4096x300]x[300x128]
  gemm_wmma_kernel<float, float><<<gemm_grid(GG, TT), 256, 0, stream>>>(
      POOL, DP, CWS, out, TT,
      DP, TT, TT,
      causal_b, nullptr, nullptr, 0);
}
